// CrossAttentionBlock_20495583937012
// MI455X (gfx1250) — matrix operand order repaired and checked by reading, not compiled, not run on hardware
//
#include <hip/hip_runtime.h>
#include <hip/hip_bf16.h>
#include <math.h>

// ---------------------------------------------------------------------------
// CDNA5 (gfx1250, wave32) cross-attention block, bf16 WMMA pipeline.
// ---------------------------------------------------------------------------

typedef __bf16 bf16_t;
typedef __attribute__((ext_vector_type(16))) __bf16 v16bf;
typedef __attribute__((ext_vector_type(8)))  __bf16 v8bf;
typedef __attribute__((ext_vector_type(4)))  __bf16 v4bf;
typedef __attribute__((ext_vector_type(8)))  float  v8f;
typedef __attribute__((ext_vector_type(4)))  float  v4f;

// D = A(16x32,bf16) * B(32x16,bf16) + C(16x16,f32)
#define WMMA_BF16(a, b, c) \
  __builtin_amdgcn_wmma_f32_16x16x32_bf16(false, (a), false, (b), (short)0, (c), false, false)

static constexpr int Bb = 2, Nn = 2048, Mm = 2048, Dd = 1024, Hh = 8, DHd = 32;
static constexpr int FFf = 4 * Dd; // 4096

__device__ inline v8bf zero8bf() {
  v8bf z;
#pragma unroll
  for (int e = 0; e < 8; ++e) z[e] = (bf16_t)0.0f;
  return z;
}
__device__ inline v8f zero8f() {
  v8f z;
#pragma unroll
  for (int e = 0; e < 8; ++e) z[e] = 0.0f;
  return z;
}

// ---------------------------------------------------------------------------
// fp32 -> bf16 conversion (weights), 4 elems/thread vectorized
// ---------------------------------------------------------------------------
__global__ __launch_bounds__(256) void cvt_f32_bf16(const float* __restrict__ in,
                                                    bf16_t* __restrict__ out, int n) {
  int i = (blockIdx.x * 256 + threadIdx.x) * 4;
  if (i + 3 < n) {
    v4f v = *(const v4f*)(in + i);
    v4bf o;
#pragma unroll
    for (int e = 0; e < 4; ++e) o[e] = (bf16_t)v[e];
    *(v4bf*)(out + i) = o;
  } else {
    for (; i < n; ++i) out[i] = (bf16_t)in[i];
  }
}

// ---------------------------------------------------------------------------
// LayerNorm over D=1024, one row per block (256 threads, 4 contiguous
// elems/thread -> b128 loads). First nx rows from x, remaining from y.
// ---------------------------------------------------------------------------
__global__ __launch_bounds__(256) void ln_kernel(const float* __restrict__ x,
                                                 const float* __restrict__ y,
                                                 const float* __restrict__ gamma,
                                                 bf16_t* __restrict__ xn,
                                                 bf16_t* __restrict__ yn, int nx) {
  const int row = blockIdx.x;
  const float* src;
  bf16_t* dst;
  if (row < nx) { src = x + (size_t)row * Dd; dst = xn + (size_t)row * Dd; }
  else          { src = y + (size_t)(row - nx) * Dd; dst = yn + (size_t)(row - nx) * Dd; }

  const int c0 = threadIdx.x * 4;
  v4f v = *(const v4f*)(src + c0);
  float s = v[0] + v[1] + v[2] + v[3];
  float ss = v[0] * v[0] + v[1] * v[1] + v[2] * v[2] + v[3] * v[3];
#pragma unroll
  for (int off = 16; off; off >>= 1) {
    s  += __shfl_xor(s,  off, 32);
    ss += __shfl_xor(ss, off, 32);
  }
  __shared__ float rs[8], rss[8];
  __shared__ float mu_s, rstd_s;
  const int w = threadIdx.x >> 5, lane = threadIdx.x & 31;
  if (lane == 0) { rs[w] = s; rss[w] = ss; }
  __syncthreads();
  if (threadIdx.x == 0) {
    float S = 0.f, SS = 0.f;
#pragma unroll
    for (int i = 0; i < 8; ++i) { S += rs[i]; SS += rss[i]; }
    float mu = S * (1.0f / Dd);
    float var = SS * (1.0f / Dd) - mu * mu;
    mu_s = mu;
    rstd_s = rsqrtf(var + 1e-5f);
  }
  __syncthreads();
  const float mu = mu_s, rstd = rstd_s;
  v4f g = *(const v4f*)(gamma + c0);
  v4bf o;
#pragma unroll
  for (int e = 0; e < 4; ++e) o[e] = (bf16_t)((v[e] - mu) * rstd * g[e]);
  *(v4bf*)(dst + c0) = o;
}

// ---------------------------------------------------------------------------
// Generic bf16 GEMM: C(Mr x Nc) = A(Mr x K, lda=K) * B(K x Nc, ldb), bf16 in,
// f32 accumulate. Register-staged double-buffered LDS pipeline.
//   non-GLU: block tile 128x128, 8 waves (4m x 2n), wave tile 32x64 (8 WMMA)
//   GLU:     block tile  64x128, 8 waves (2m x 4n), wave tile 32x32, dual B
// Epilogues: 0 bf16 store | 1 f32 store | 2 f32 accumulate |
//            3 SwiGLU: h = silu(B[:, n+gateOff]col) * B[:, n]col, bf16 store
// ---------------------------------------------------------------------------
enum : int { EPI_BF16 = 0, EPI_F32 = 1, EPI_F32ACC = 2, EPI_GLU = 3 };

template <int EPI>
__global__ __launch_bounds__(256) void gemm_bf16(const bf16_t* __restrict__ A,
                                                 const bf16_t* __restrict__ Bm,
                                                 float* __restrict__ Cf,
                                                 bf16_t* __restrict__ Cb, int Mr, int Nc,
                                                 int K, int ldb, int gateOff) {
  constexpr bool GLU = (EPI == EPI_GLU);
  constexpr int BM = GLU ? 64 : 128;
  constexpr int BN = 128, KC = 32;
  constexpr int WNW = GLU ? 4 : 2;             // waves along N
  constexpr int SUBM = 2;                      // 16-row tiles per wave (32 rows)
  constexpr int SUBN = (BN / WNW) / 16;        // GLU: 2, else 4
  constexpr int NA = (BM * KC) / (256 * 8);    // v8bf A-chunks per thread (1|2)
  constexpr int NB = (KC * BN) / (256 * 8);    // v8bf B-chunks per thread (2)

  __shared__ alignas(32) bf16_t As[BM][KC];
  __shared__ alignas(32) bf16_t Bs[KC][BN];
  __shared__ alignas(32) bf16_t Bs2[GLU ? KC : 1][GLU ? BN : 1];

  const int t = threadIdx.x;
  const int m0 = blockIdx.y * BM;
  const int n0 = blockIdx.x * BN;
  const int lane = t & 31, w = t >> 5;
  const int wm = w / WNW, wn = w % WNW;
  const int lr = lane & 15;
  const int kg = (lane >> 4) * 8; // A-fragment K interleave base

  v8f acc[SUBM][SUBN], accg[GLU ? SUBM : 1][GLU ? SUBN : 1];
#pragma unroll
  for (int i = 0; i < SUBM; ++i)
#pragma unroll
    for (int j = 0; j < SUBN; ++j) {
      acc[i][j] = zero8f();
      if constexpr (GLU) accg[i][j] = zero8f();
    }

  v8bf avr[NA], bvr[NB], gvr[NB];

  auto loadAB = [&](int k0) {
#pragma unroll
    for (int i = 0; i < NA; ++i) {
      const int idx = t + i * 256;
      const int row = m0 + (idx >> 2);
      const int col = k0 + (idx & 3) * 8;
      avr[i] = (row < Mr) ? *(const v8bf*)(A + (size_t)row * K + col) : zero8bf();
    }
#pragma unroll
    for (int i = 0; i < NB; ++i) {
      const int idx = t + i * 256;
      const int row = k0 + (idx >> 4);
      const int col = n0 + (idx & 15) * 8;
      const bf16_t* bp = Bm + (size_t)row * ldb + col;
      bvr[i] = (col < Nc) ? *(const v8bf*)bp : zero8bf();
      if constexpr (GLU) gvr[i] = (col < Nc) ? *(const v8bf*)(bp + gateOff) : zero8bf();
    }
  };
  auto depositAB = [&]() {
#pragma unroll
    for (int i = 0; i < NA; ++i) {
      const int idx = t + i * 256;
      *(v8bf*)&As[idx >> 2][(idx & 3) * 8] = avr[i];
    }
#pragma unroll
    for (int i = 0; i < NB; ++i) {
      const int idx = t + i * 256;
      *(v8bf*)&Bs[idx >> 4][(idx & 15) * 8] = bvr[i];
      if constexpr (GLU) *(v8bf*)&Bs2[idx >> 4][(idx & 15) * 8] = gvr[i];
    }
  };

  // pipeline prologue
  loadAB(0);
  depositAB();
  __syncthreads();

  for (int k0 = 0; k0 < K; k0 += KC) {
    const bool last = (k0 + KC >= K);
    if (!last) loadAB(k0 + KC); // global loads in flight while computing
    if (k0 + 2 * KC < K) {      // -> global_prefetch_b8
      __builtin_prefetch(A + (size_t)(m0 + (t >> 2)) * K + (k0 + 2 * KC) + (t & 3) * 8, 0, 1);
      __builtin_prefetch(Bm + (size_t)(k0 + 2 * KC + (t >> 4)) * ldb + n0 + (t & 15) * 8, 0, 1);
    }

    // ---- fragments from LDS ----
    v16bf afr[SUBM], bfr[SUBN], bfr2[GLU ? SUBN : 1];
#pragma unroll
    for (int sm = 0; sm < SUBM; ++sm) {
      const int row = wm * (SUBM * 16) + sm * 16 + lr;
      v8bf lo = *(const v8bf*)&As[row][kg];
      v8bf hi = *(const v8bf*)&As[row][16 + kg];
      afr[sm] = __builtin_shufflevector(lo, hi, 0, 1, 2, 3, 4, 5, 6, 7, 8, 9, 10, 11, 12, 13, 14, 15);
    }
#pragma unroll
    for (int sn = 0; sn < SUBN; ++sn) {
      const int col = wn * (SUBN * 16) + sn * 16;
      for (int s = 0; s < 16; ++s) { // B layout: N = lane%16, K table order
        const int kk = (s < 8) ? (kg + s) : (16 + kg + s - 8);
        bfr[sn][s] = Bs[kk][col + lr];
        if constexpr (GLU) bfr2[sn][s] = Bs2[kk][col + lr];
      }
    }
#pragma unroll
    for (int sm = 0; sm < SUBM; ++sm)
#pragma unroll
      for (int sn = 0; sn < SUBN; ++sn) {
        acc[sm][sn] = WMMA_BF16(afr[sm], bfr[sn], acc[sm][sn]);
        if constexpr (GLU) accg[sm][sn] = WMMA_BF16(afr[sm], bfr2[sn], accg[sm][sn]);
      }
    __syncthreads();
    if (!last) {
      depositAB();
      __syncthreads();
    }
  }

  // ---- epilogue (C layout: col = lane%16, row = r + 8*(lane/16)) ----
#pragma unroll
  for (int sm = 0; sm < SUBM; ++sm)
#pragma unroll
    for (int sn = 0; sn < SUBN; ++sn)
#pragma unroll
      for (int r = 0; r < 8; ++r) {
        const int row = m0 + wm * (SUBM * 16) + sm * 16 + (lane >> 4) * 8 + r;
        const int col = n0 + wn * (SUBN * 16) + sn * 16 + lr;
        if (row < Mr && col < Nc) {
          const float vv = acc[sm][sn][r];
          if constexpr (EPI == EPI_BF16) {
            Cb[(size_t)row * Nc + col] = (bf16_t)vv;
          } else if constexpr (EPI == EPI_F32) {
            Cf[(size_t)row * Nc + col] = vv;
          } else if constexpr (EPI == EPI_F32ACC) {
            Cf[(size_t)row * Nc + col] += vv;
          } else { // SwiGLU: silu(gate) * val
            const float g = accg[sm][sn][r];
            Cb[(size_t)row * Nc + col] = (bf16_t)(vv * g / (1.0f + __expf(-g)));
          }
        }
      }
}

// ---------------------------------------------------------------------------
// Flash attention, one wave per 16 query rows of one (b,h).
// Computes S^T = K_tile(16x32) * q^T(32x16): each LANE owns one query row
// (i = lane%16) so softmax max/sum are per-lane, and the S^T C-fragment
// layout (j = 8*(lane/16)+r) is exactly the A-fragment layout for P @ V.
// 1/sqrt(DH) folded in here.
// ---------------------------------------------------------------------------
__global__ __launch_bounds__(256) void attn_kernel(const bf16_t* __restrict__ q,  // (B,N,H*DH)
                                                   const bf16_t* __restrict__ kv, // (B,M,2*DH)
                                                   bf16_t* __restrict__ out) {    // (B,N,H*DH)
  const int lane = threadIdx.x & 31;
  const int gw = blockIdx.x * 8 + (threadIdx.x >> 5);
  const int ntiles = Nn / 16;
  const int it = gw % ntiles;
  const int bh = gw / ntiles;
  const int h = bh % Hh;
  const int b = bh / Hh;
  const int i0 = it * 16;
  const int lr = lane & 15;
  const int kg = (lane >> 4) * 8;
  const float scale = 0.17677669529663689f; // DH^-0.5

  // q^T B-fragment: column i = lane%16, elements = d in K-interleaved chunks
  v16bf bq;
  {
    const bf16_t* qp = q + ((size_t)b * Nn + i0 + lr) * (Hh * DHd) + h * DHd;
    v8bf qlo = *(const v8bf*)(qp + kg);
    v8bf qhi = *(const v8bf*)(qp + 16 + kg);
    bq = __builtin_shufflevector(qlo, qhi, 0, 1, 2, 3, 4, 5, 6, 7, 8, 9, 10, 11, 12, 13, 14, 15);
  }

  float mrow = -__builtin_inff();
  float ssum = 0.f;
  v8f o0 = zero8f(), o1 = zero8f();
  const bf16_t* kvb = kv + (size_t)b * Mm * (2 * DHd);

  for (int j0 = 0; j0 < Mm; j0 += 32) {
    // K A-fragments: rows j0+lr and j0+16+lr, K-interleaved d chunks
    const bf16_t* kp0 = kvb + (size_t)(j0 + lr) * (2 * DHd);
    const bf16_t* kp1 = kvb + (size_t)(j0 + 16 + lr) * (2 * DHd);
    v8bf lo0 = *(const v8bf*)(kp0 + kg);
    v8bf hi0 = *(const v8bf*)(kp0 + 16 + kg);
    v8bf lo1 = *(const v8bf*)(kp1 + kg);
    v8bf hi1 = *(const v8bf*)(kp1 + 16 + kg);
    v16bf a0 = __builtin_shufflevector(lo0, hi0, 0, 1, 2, 3, 4, 5, 6, 7, 8, 9, 10, 11, 12, 13, 14, 15);
    v16bf a1 = __builtin_shufflevector(lo1, hi1, 0, 1, 2, 3, 4, 5, 6, 7, 8, 9, 10, 11, 12, 13, 14, 15);

    v8f s0 = WMMA_BF16(a0, bq, zero8f()); // S^T rows j0+kg+r
    v8f s1 = WMMA_BF16(a1, bq, zero8f()); // S^T rows j0+16+kg+r

    // --- online softmax (per-lane row i = lr) ---
    float cmax = -__builtin_inff();
#pragma unroll
    for (int e = 0; e < 8; ++e) cmax = fmaxf(cmax, fmaxf(s0[e], s1[e]));
    cmax = fmaxf(cmax, __shfl_xor(cmax, 16, 32));
    const float mnew = fmaxf(mrow, cmax * scale);
    const float alpha = __expf(mrow - mnew);

    v16bf pf;
    float psum = 0.f;
#pragma unroll
    for (int e = 0; e < 8; ++e) {
      float p = __expf(s0[e] * scale - mnew);
      pf[e] = (bf16_t)p;
      psum += p;
    }
#pragma unroll
    for (int e = 0; e < 8; ++e) {
      float p = __expf(s1[e] * scale - mnew);
      pf[8 + e] = (bf16_t)p;
      psum += p;
    }
    ssum = ssum * alpha + psum;
    mrow = mnew;

    // rescale O by alpha of its row: row i = r + 8*(lane/16); alpha lives on lane i
#pragma unroll
    for (int r = 0; r < 8; ++r) {
      float ar_ = __shfl(alpha, r + ((lane >> 4) << 3), 32);
      o0[r] *= ar_;
      o1[r] *= ar_;
    }

    // V B-fragments: column d = lane%16 (+16), elements = j in K-interleaved order
    const bf16_t* vp = kvb + (size_t)j0 * (2 * DHd) + DHd + lr;
    v16bf bv0, bv1;
    for (int s = 0; s < 16; ++s) {
      const int jj = (s < 8) ? (kg + s) : (16 + kg + s - 8);
      bv0[s] = vp[(size_t)jj * (2 * DHd)];
      bv1[s] = vp[(size_t)jj * (2 * DHd) + 16];
    }
    o0 = WMMA_BF16(pf, bv0, o0);
    o1 = WMMA_BF16(pf, bv1, o1);
  }

  const float stot = ssum + __shfl_xor(ssum, 16, 32);
  const float inv = 1.0f / stot; // valid on lane i = lr
  bf16_t* op = out + ((size_t)b * Nn + i0) * (Hh * DHd) + h * DHd;
#pragma unroll
  for (int r = 0; r < 8; ++r) {
    const int irel = r + ((lane >> 4) << 3);
    const float invr = __shfl(inv, irel, 32);
    op[(size_t)irel * (Hh * DHd) + lr] = (bf16_t)(o0[r] * invr);
    op[(size_t)irel * (Hh * DHd) + 16 + lr] = (bf16_t)(o1[r] * invr);
  }
}

// ---------------------------------------------------------------------------
// Launch
// ---------------------------------------------------------------------------
extern "C" void kernel_launch(void* const* d_in, const int* in_sizes, int n_in,
                              void* d_out, int out_size, void* d_ws, size_t ws_size,
                              hipStream_t stream) {
  (void)in_sizes; (void)n_in; (void)out_size; (void)ws_size;
  const float* x     = (const float*)d_in[0];
  const float* y     = (const float*)d_in[1];
  const float* gamma = (const float*)d_in[2];
  const float* w_q   = (const float*)d_in[3];
  const float* w_kv  = (const float*)d_in[4];
  const float* w_out = (const float*)d_in[5];
  const float* w_ff1 = (const float*)d_in[6];
  const float* w_ff2 = (const float*)d_in[7];
  float* out = (float*)d_out;

  const int RX = Bb * Nn; // 4096 x-rows
  const int RY = Bb * Mm; // 4096 y-rows

  // workspace carve (bf16, 256B aligned)
  char* p = (char*)d_ws;
  auto alloc = [&](size_t nelem) -> bf16_t* {
    bf16_t* r = (bf16_t*)p;
    p += (nelem * sizeof(bf16_t) + 255) & ~(size_t)255;
    return r;
  };
  bf16_t* wq_b  = alloc((size_t)Dd * Hh * DHd);
  bf16_t* wkv_b = alloc((size_t)Dd * 2 * DHd);
  bf16_t* wo_b  = alloc((size_t)Hh * DHd * Dd);
  bf16_t* wf1_b = alloc((size_t)Dd * 2 * FFf);
  bf16_t* wf2_b = alloc((size_t)FFf * Dd);
  bf16_t* xn    = alloc((size_t)RX * Dd);
  bf16_t* yn    = alloc((size_t)RY * Dd);
  bf16_t* qb    = alloc((size_t)RX * Hh * DHd);
  bf16_t* kvb   = alloc((size_t)RY * 2 * DHd);
  bf16_t* ao    = alloc((size_t)RX * Hh * DHd);
  bf16_t* hb    = alloc((size_t)RX * FFf);

  auto cvt = [&](const float* s, bf16_t* d, size_t n) {
    cvt_f32_bf16<<<(unsigned)((n / 4 + 255) / 256), 256, 0, stream>>>(s, d, (int)n);
  };
  cvt(w_q,   wq_b,  (size_t)Dd * Hh * DHd);
  cvt(w_kv,  wkv_b, (size_t)Dd * 2 * DHd);
  cvt(w_out, wo_b,  (size_t)Hh * DHd * Dd);
  cvt(w_ff1, wf1_b, (size_t)Dd * 2 * FFf);
  cvt(w_ff2, wf2_b, (size_t)FFf * Dd);

  ln_kernel<<<RX + RY, 256, 0, stream>>>(x, y, gamma, xn, yn, RX);

  // q = xn @ w_q  (4096x1024 * 1024x256 -> bf16)
  gemm_bf16<EPI_BF16><<<dim3((Hh * DHd + 127) / 128, RX / 128), 256, 0, stream>>>(
      xn, wq_b, nullptr, qb, RX, Hh * DHd, Dd, Hh * DHd, 0);
  // kv = yn @ w_kv (4096x1024 * 1024x64 -> bf16)
  gemm_bf16<EPI_BF16><<<dim3(1, RY / 128), 256, 0, stream>>>(
      yn, wkv_b, nullptr, kvb, RY, 2 * DHd, Dd, 2 * DHd, 0);

  // attention: one wave per 16 q rows per (b,h)
  attn_kernel<<<(Bb * Hh * (Nn / 16)) / 8, 256, 0, stream>>>(qb, kvb, ao);

  // out = ao @ w_out (4096x256 * 256x1024 -> f32, overwrites poison)
  gemm_bf16<EPI_F32><<<dim3(Dd / 128, RX / 128), 256, 0, stream>>>(
      ao, wo_b, out, nullptr, RX, Dd, Hh * DHd, Dd, 0);

  // h = silu(xn@w_ff1[:,FF:]) * (xn@w_ff1[:,:FF]) (fused SwiGLU, bf16)
  gemm_bf16<EPI_GLU><<<dim3(FFf / 128, RX / 64), 256, 0, stream>>>(
      xn, wf1_b, nullptr, hb, RX, FFf, Dd, 2 * FFf, FFf);

  // out += h @ w_ff2 (4096x4096 * 4096x1024, f32 accumulate)
  gemm_bf16<EPI_F32ACC><<<dim3(Dd / 128, RX / 128), 256, 0, stream>>>(
      hb, wf2_b, out, nullptr, RX, Dd, FFf, Dd, 0);
}
